// Model_36532991819814
// MI455X (gfx1250) — matrix operand order repaired and checked by reading, not compiled, not run on hardware
//
#include <hip/hip_runtime.h>
#include <hip/hip_bf16.h>

// ---------------- problem constants ----------------
#define BATCH 4096
#define SEQ   50
#define DH    47      // feature dim
#define XCH   28      // x channels: 1 id + 27 feats
#define EMBD  20
#define VSZ   32768
#define ZPAD  16      // z padded to 16 (zeros beyond 11)
#define KPAD  32      // WMMA K

typedef __attribute__((ext_vector_type(16))) _Float16 v16h;
typedef __attribute__((ext_vector_type(8)))  _Float16 v8h;
typedef __attribute__((ext_vector_type(8)))  float    v8f;

// ---------------- K0: pre-pack fc4_w into f16 B-fragment layout ----------------
// whB[col][k], k=0..31 contiguous halves; rows k>=11 are zero.  2 MB, L2-hot.

__global__ __launch_bounds__(256) void pack_w_kernel(const float* __restrict__ W,
                                                     _Float16* __restrict__ whB) {
    const int col = blockIdx.x * blockDim.x + threadIdx.x;
    if (col >= VSZ) return;
    _Float16* dst = whB + (size_t)col * KPAD;
    #pragma unroll
    for (int k = 0; k < KPAD; ++k)
        dst[k] = (k < 11) ? (_Float16)W[(size_t)k * VSZ + col] : (_Float16)0.0f;
}

// ---------------- K1: per-row attention head + SE + MLP -> z[11] (f16, padded) ----------------

__device__ __forceinline__ void dense_layer(const float* in, float* out,
                                            const float* W, const float* b,
                                            int nin, int nout, int t) {
    if (t < nout) {
        float acc = b[t];
        for (int i = 0; i < nin; ++i) acc += in[i] * W[i * nout + t];
        out[t] = fmaxf(acc, 0.0f);   // all layers in the chain are ReLU
    }
}

__global__ __launch_bounds__(128) void seq_head_kernel(
    const float* __restrict__ x, const int* __restrict__ mask, const float* __restrict__ embed,
    const float* __restrict__ Wq_w, const float* __restrict__ Wq_b,
    const float* __restrict__ Wk_w, const float* __restrict__ Wk_b,
    const float* __restrict__ Wv_w, const float* __restrict__ Wv_b,
    const float* __restrict__ se1_w, const float* __restrict__ se1_b,
    const float* __restrict__ se2_w, const float* __restrict__ se2_b,
    const float* __restrict__ fc1_w,  const float* __restrict__ fc1_b,
    const float* __restrict__ fc12_w, const float* __restrict__ fc12_b,
    const float* __restrict__ fc2_w,  const float* __restrict__ fc2_b,
    const float* __restrict__ fc22_w, const float* __restrict__ fc22_b,
    const float* __restrict__ fc3_w,  const float* __restrict__ fc3_b,
    const float* __restrict__ fc32_w, const float* __restrict__ fc32_b,
    _Float16* __restrict__ zout)
{
    const int b = blockIdx.x;
    const int t = threadIdx.x;

    __shared__ float h_sh[SEQ][DH];
    __shared__ float q_sh[DH];
    __shared__ float wqk[DH];
    __shared__ float hb[DH];
    __shared__ float sc[SEQ];
    __shared__ float at[SEQ];
    __shared__ float bufA[2 * DH];
    __shared__ float bufB[2 * DH];
    __shared__ float pm[6], gate[6], seh[2];
    __shared__ float scal[3];          // [0]=q.bk  [1]=max  [2]=1/sum
    __shared__ int   len_sh;

    if (t == 0) {
        int L = 0;
        for (int s = 0; s < SEQ; ++s) L += mask[b * SEQ + s];
        len_sh = L;
    }
    // build h = [embed[id] (20) | feats (27)]
    for (int idx = t; idx < SEQ * DH; idx += blockDim.x) {
        int s = idx / DH, d = idx % DH;
        float v;
        if (d < EMBD) {
            int id = (int)x[((size_t)b * SEQ + s) * XCH];
            v = embed[(size_t)id * EMBD + d];
        } else {
            v = x[((size_t)b * SEQ + s) * XCH + 1 + (d - EMBD)];
        }
        h_sh[s][d] = v;
    }
    __syncthreads();

    const int len = len_sh;
    const int ql  = len - 1;

    // q = h[ql] @ Wq + bq
    if (t < DH) {
        float acc = Wq_b[t];
        for (int i = 0; i < DH; ++i) acc += h_sh[ql][i] * Wq_w[i * DH + t];
        q_sh[t] = acc;
    }
    __syncthreads();

    // wqk = Wk @ q  (score_s = h_s . wqk + q.bk)
    if (t < DH) {
        float acc = 0.0f;
        for (int j = 0; j < DH; ++j) acc += Wk_w[t * DH + j] * q_sh[j];
        wqk[t] = acc;
    } else if (t == DH) {
        float acc = 0.0f;
        for (int j = 0; j < DH; ++j) acc += q_sh[j] * Wk_b[j];
        scal[0] = acc;
    }
    __syncthreads();

    if (t < SEQ) {
        float acc = scal[0];
        for (int i = 0; i < DH; ++i) acc += h_sh[t][i] * wqk[i];
        sc[t] = (t < len - 1) ? acc : -1.0e30f;
    }
    __syncthreads();
    if (t == 0) {
        float m = -1.0e30f;
        for (int s = 0; s < SEQ; ++s) m = fmaxf(m, sc[s]);
        scal[1] = m;
    }
    __syncthreads();
    if (t < SEQ) at[t] = __expf(sc[t] - scal[1]);
    __syncthreads();
    if (t == 0) {
        float s0 = 0.0f;
        for (int s = 0; s < SEQ; ++s) s0 += at[s];
        scal[2] = 1.0f / s0;
    }
    __syncthreads();
    if (t < SEQ) at[t] *= scal[2];
    __syncthreads();

    // hb = sum_s attn_s * h_s ; prev = tanh(hb @ Wv + bv)  (sum attn == 1)
    if (t < DH) {
        float acc = 0.0f;
        for (int s = 0; s < SEQ; ++s) acc += at[s] * h_sh[s][t];
        hb[t] = acc;
    }
    __syncthreads();
    if (t < DH) {
        float acc = Wv_b[t];
        for (int i = 0; i < DH; ++i) acc += hb[i] * Wv_w[i * DH + t];
        bufA[t] = tanhf(acc);
    } else if (t < 2 * DH) {
        bufA[t] = h_sh[SEQ - 1][t - DH];   // next_item
    }
    __syncthreads();

    // SE block over 6 parts of out[94]
    const int offs[7] = {0, 20, 26, 47, 67, 73, 94};
    if (t < 6) {
        float acc = 0.0f;
        for (int i = offs[t]; i < offs[t + 1]; ++i) acc += bufA[i];
        pm[t] = acc / (float)(offs[t + 1] - offs[t]);
    }
    __syncthreads();
    if (t < 2) {
        float acc = se1_b[t];
        for (int i = 0; i < 6; ++i) acc += pm[i] * se1_w[i * 2 + t];
        seh[t] = fmaxf(acc, 0.0f);
    }
    __syncthreads();
    if (t < 6) {
        float acc = se2_b[t];
        for (int i = 0; i < 2; ++i) acc += seh[i] * se2_w[i * 6 + t];
        gate[t] = 1.0f / (1.0f + __expf(-acc));
    }
    __syncthreads();
    if (t < 2 * DH) {
        int p = 0;
        #pragma unroll
        for (int i = 1; i < 6; ++i) p += (t >= offs[i]) ? 1 : 0;
        bufA[t] *= gate[p];
    }
    __syncthreads();

    // MLP chain (all ReLU): 94->47->47->23->23->11->11
    dense_layer(bufA, bufB, fc1_w,  fc1_b,  94, 47, t); __syncthreads();
    dense_layer(bufB, bufA, fc12_w, fc12_b, 47, 47, t); __syncthreads();
    dense_layer(bufA, bufB, fc2_w,  fc2_b,  47, 23, t); __syncthreads();
    dense_layer(bufB, bufA, fc22_w, fc22_b, 23, 23, t); __syncthreads();
    dense_layer(bufA, bufB, fc3_w,  fc3_b,  23, 11, t); __syncthreads();
    dense_layer(bufB, bufA, fc32_w, fc32_b, 11, 11, t); __syncthreads();

    if (t < ZPAD)
        zout[(size_t)b * ZPAD + t] = (_Float16)((t < 11) ? bufA[t] : 0.0f);
}

// ---------------- K2: logits = z @ fc4_w + b, softmax over V, via WMMA ----------------
// One WG per 16 batch rows; 16 waves/WG, each wave owns V/16 = 2048 columns.
// B fragments come branch-free from the pre-packed f16 whB buffer:
//   lanes 0-15 load K0-7 & K16-23 of their column, lanes 16-31 load K8-15 & K24-31.

__device__ __forceinline__ v16h load_B_frag(const _Float16* __restrict__ whB,
                                            int col, int halfId) {
    const _Float16* p = whB + (size_t)col * KPAD + halfId * 8;
    v8h b0 = *(const v8h*)(p);
    v8h b1 = *(const v8h*)(p + 16);
    v16h Bv;
    #pragma unroll
    for (int e = 0; e < 8; ++e) { Bv[e] = b0[e]; Bv[e + 8] = b1[e]; }
    return Bv;
}

__global__ __launch_bounds__(512) void fc4_softmax_kernel(
    const _Float16* __restrict__ z, const _Float16* __restrict__ whB,
    const float* __restrict__ bias, float* __restrict__ out)
{
    const int rowBase = blockIdx.x * 16;
    const int tid  = threadIdx.x;
    const int lane = tid & 31;
    const int wave = tid >> 5;           // 0..15
    const int halfId = lane >> 4;        // 0 or 1
    const int l16 = lane & 15;

    // A fragment: 16x32 f16. Lanes 0-15: M=lane, K=0-7 (elems 0-7) & K=16-23 (elems 8-15);
    // lanes 16-31: M=lane-16, K=8-15 & K=24-31. K>=11 zero (z pre-padded; K>=16 forced 0).
    v16h A;
    {
        const int m = l16, kb = halfId * 8;
        v8h lo = *(const v8h*)(z + (size_t)(rowBase + m) * ZPAD + kb);
        #pragma unroll
        for (int e = 0; e < 8; ++e) { A[e] = lo[e]; A[e + 8] = (_Float16)0.0f; }
    }

    const int colsPerWave = VSZ / 16;    // 2048
    const int cLo = wave * colsPerWave;
    const int cHi = cLo + colsPerWave;

    __shared__ float smax[16][16];
    __shared__ float ssum[16][16];
    __shared__ float fm_sh[16], fs_sh[16];

    // ---- pass 1: per-row max ----
    float pmax[8];
    #pragma unroll
    for (int r = 0; r < 8; ++r) pmax[r] = -3.0e38f;

    for (int c0 = cLo; c0 < cHi; c0 += 16) {
        const int col = c0 + l16;
        v16h Bv = load_B_frag(whB, col, halfId);
        v8f  C = {};
        C = __builtin_amdgcn_wmma_f32_16x16x32_f16(false, A, false, Bv, (short)0, C, false, false);
        const float bcol = bias[col];
        #pragma unroll
        for (int r = 0; r < 8; ++r) pmax[r] = fmaxf(pmax[r], C[r] + bcol);
    }
    #pragma unroll
    for (int r = 0; r < 8; ++r) {
        float v = pmax[r];
        v = fmaxf(v, __shfl_xor(v, 1));
        v = fmaxf(v, __shfl_xor(v, 2));
        v = fmaxf(v, __shfl_xor(v, 4));
        v = fmaxf(v, __shfl_xor(v, 8));
        pmax[r] = v;
    }
    if (l16 == 0) {
        #pragma unroll
        for (int r = 0; r < 8; ++r) smax[wave][halfId * 8 + r] = pmax[r];
    }
    __syncthreads();
    if (tid < 16) {
        float m0 = -3.0e38f;
        for (int w = 0; w < 16; ++w) m0 = fmaxf(m0, smax[w][tid]);
        fm_sh[tid] = m0;
    }
    __syncthreads();
    float fm[8];
    #pragma unroll
    for (int r = 0; r < 8; ++r) fm[r] = fm_sh[halfId * 8 + r];

    // ---- pass 2: per-row sum of exp ----
    float psum[8];
    #pragma unroll
    for (int r = 0; r < 8; ++r) psum[r] = 0.0f;

    for (int c0 = cLo; c0 < cHi; c0 += 16) {
        const int col = c0 + l16;
        v16h Bv = load_B_frag(whB, col, halfId);
        v8f  C = {};
        C = __builtin_amdgcn_wmma_f32_16x16x32_f16(false, A, false, Bv, (short)0, C, false, false);
        const float bcol = bias[col];
        #pragma unroll
        for (int r = 0; r < 8; ++r) psum[r] += __expf(C[r] + bcol - fm[r]);
    }
    #pragma unroll
    for (int r = 0; r < 8; ++r) {
        float v = psum[r];
        v += __shfl_xor(v, 1);
        v += __shfl_xor(v, 2);
        v += __shfl_xor(v, 4);
        v += __shfl_xor(v, 8);
        psum[r] = v;
    }
    if (l16 == 0) {
        #pragma unroll
        for (int r = 0; r < 8; ++r) ssum[wave][halfId * 8 + r] = psum[r];
    }
    __syncthreads();
    if (tid < 16) {
        float s0 = 0.0f;
        for (int w = 0; w < 16; ++w) s0 += ssum[w][tid];
        fs_sh[tid] = 1.0f / s0;
    }
    __syncthreads();
    float fs[8];
    #pragma unroll
    for (int r = 0; r < 8; ++r) fs[r] = fs_sh[halfId * 8 + r];

    // ---- pass 3: recompute and write normalized probabilities ----
    for (int c0 = cLo; c0 < cHi; c0 += 16) {
        const int col = c0 + l16;
        v16h Bv = load_B_frag(whB, col, halfId);
        v8f  C = {};
        C = __builtin_amdgcn_wmma_f32_16x16x32_f16(false, A, false, Bv, (short)0, C, false, false);
        const float bcol = bias[col];
        #pragma unroll
        for (int r = 0; r < 8; ++r) {
            const int row = rowBase + halfId * 8 + r;
            out[(size_t)row * VSZ + col] = __expf(C[r] + bcol - fm[r]) * fs[r];
        }
    }
}

// ---------------- launch ----------------
extern "C" void kernel_launch(void* const* d_in, const int* in_sizes, int n_in,
                              void* d_out, int out_size, void* d_ws, size_t ws_size,
                              hipStream_t stream) {
    (void)in_sizes; (void)n_in; (void)out_size; (void)ws_size;
    const float* x      = (const float*)d_in[0];
    const int*   mask   = (const int*)  d_in[1];
    const float* embed  = (const float*)d_in[2];
    const float* Wq_w   = (const float*)d_in[3];
    const float* Wq_b   = (const float*)d_in[4];
    const float* Wk_w   = (const float*)d_in[5];
    const float* Wk_b   = (const float*)d_in[6];
    const float* Wv_w   = (const float*)d_in[7];
    const float* Wv_b   = (const float*)d_in[8];
    const float* se1_w  = (const float*)d_in[9];
    const float* se1_b  = (const float*)d_in[10];
    const float* se2_w  = (const float*)d_in[11];
    const float* se2_b  = (const float*)d_in[12];
    const float* fc1_w  = (const float*)d_in[13];
    const float* fc1_b  = (const float*)d_in[14];
    const float* fc12_w = (const float*)d_in[15];
    const float* fc12_b = (const float*)d_in[16];
    const float* fc2_w  = (const float*)d_in[17];
    const float* fc2_b  = (const float*)d_in[18];
    const float* fc22_w = (const float*)d_in[19];
    const float* fc22_b = (const float*)d_in[20];
    const float* fc3_w  = (const float*)d_in[21];
    const float* fc3_b  = (const float*)d_in[22];
    const float* fc32_w = (const float*)d_in[23];
    const float* fc32_b = (const float*)d_in[24];
    const float* fc4_w  = (const float*)d_in[25];
    const float* fc4_b  = (const float*)d_in[26];

    // workspace layout: [0,128KB) z (f16, BATCH x 16) ; [128KB, 128KB+2MB) packed fc4_w
    _Float16* zbuf = (_Float16*)d_ws;
    _Float16* whB  = (_Float16*)((char*)d_ws + (size_t)BATCH * ZPAD * sizeof(_Float16));

    pack_w_kernel<<<VSZ / 256, 256, 0, stream>>>(fc4_w, whB);

    seq_head_kernel<<<BATCH, 128, 0, stream>>>(
        x, mask, embed, Wq_w, Wq_b, Wk_w, Wk_b, Wv_w, Wv_b,
        se1_w, se1_b, se2_w, se2_b,
        fc1_w, fc1_b, fc12_w, fc12_b, fc2_w, fc2_b, fc22_w, fc22_b,
        fc3_w, fc3_b, fc32_w, fc32_b, zbuf);

    fc4_softmax_kernel<<<BATCH / 16, 512, 0, stream>>>(
        zbuf, whB, fc4_b, (float*)d_out);
}
